// SimpleRNN_75265006895141
// MI455X (gfx1250) — compile-verified, matrix operand order checked by reading, not run on hardware
//
#include <hip/hip_runtime.h>
#include <stdint.h>

// SimpleRNN for MI455X (gfx1250, wave32, WMMA).
// Pipeline: pack f32->f16 | x_proj GEMM (WMMA) | sequential scan (WMMA, W_hh in VGPRs)
//           | logits GEMM (WMMA, async global->LDS A-staging, NT output stores)

#define VOCAB   16000
#define EMBED   256
#define HIDDEN  512
#define BATCH   8
#define SEQ     2048
#define M_TOTAL (BATCH * SEQ)   // 16384 rows = b*SEQ + t

typedef __attribute__((ext_vector_type(16))) _Float16 v16h;
typedef __attribute__((ext_vector_type(8)))  float    v8f;
typedef __attribute__((ext_vector_type(4)))  int      v4i;

union FragU {          // one WMMA 16-bit operand fragment per lane: 16 halves = 2x b128
    v16h  h;
    uint4 u[2];
};

static __device__ __forceinline__ v8f wmma_f16(v16h a, v16h b, v8f c) {
    // D = A(16x32 f16) x B(32x16 f16) + C(16x16 f32)
    return __builtin_amdgcn_wmma_f32_16x16x32_f16(false, a, false, b, (short)0, c, false, false);
}

// ---- gfx1250 async global->LDS copy (ASYNCcnt path), with safe fallback ----
#if defined(__has_builtin)
#if __has_builtin(__builtin_amdgcn_global_load_async_to_lds_b128)
#define HAVE_ASYNC_LDS 1
#endif
#endif

static __device__ __forceinline__ void copy16_g2lds(const void* g, void* l) {
#ifdef HAVE_ASYNC_LDS
    // Builtin expects vector-typed pointers (param0: 'int __vector(4)*' per clang);
    // use concrete int4* in the proper address spaces (global src, LDS dst).
    __builtin_amdgcn_global_load_async_to_lds_b128(
        (__attribute__((address_space(1))) v4i*)g,
        (__attribute__((address_space(3))) v4i*)l, 0, 0);
#else
    *(uint4*)l = *(const uint4*)g;
#endif
}

static __device__ __forceinline__ void wait_async_copies() {
#ifdef HAVE_ASYNC_LDS
#if __has_builtin(__builtin_amdgcn_s_wait_asynccnt)
    __builtin_amdgcn_s_wait_asynccnt(0);
#else
    asm volatile("s_wait_asynccnt 0x0" ::: "memory");
#endif
#endif
}

// ---------------------------------------------------------------------------
// Kernel 0: pack pairs of consecutive f32 into u32 of two f16 (row-major safe)
// ---------------------------------------------------------------------------
__global__ void pack_f32_to_f16(const float* __restrict__ in,
                                uint32_t* __restrict__ out, int npairs) {
    int i = blockIdx.x * blockDim.x + threadIdx.x;
    if (i < npairs) {
        union { _Float16 h[2]; uint32_t u; } p;
        p.h[0] = (_Float16)in[2 * i];
        p.h[1] = (_Float16)in[2 * i + 1];
        out[i] = p.u;
    }
}

// ---------------------------------------------------------------------------
// Kernel 1: x_proj[m][n] = sum_k emb[x[m]][k] * W_ih[n][k] + b_ih[n]
// Block = 256 thr (8 waves). Block owns one 16-row M tile; wave w owns cols [64w,64w+64).
// Embedding gather is fused into the A-fragment loads (token row -> b128 loads).
// ---------------------------------------------------------------------------
__global__ __launch_bounds__(256) void xproj_kernel(
    const int* __restrict__ x,
    const uint32_t* __restrict__ embh,   // [VOCAB][EMBED/2] packed f16
    const uint32_t* __restrict__ wihh,   // [HIDDEN][EMBED/2] packed f16
    const float* __restrict__ b_ih,
    float* __restrict__ xproj)           // [M_TOTAL][HIDDEN] f32
{
    const int lane = threadIdx.x & 31;
    const int wave = threadIdx.x >> 5;          // 0..7
    const int lr   = lane & 15;
    const int hi   = lane >> 4;                 // K-half select per ISA A/B layout
    const int m0   = blockIdx.x * 16;
    const int n0   = wave * 64;

    const int   token = x[m0 + lr];             // A row m = m0 + (lane&15) (both lane halves)
    const char* arow  = (const char*)embh + (size_t)token * (EMBED * 2);

    v8f acc[4] = {};
#pragma unroll
    for (int kf = 0; kf < EMBED / 32; ++kf) {
        FragU a;
        const char* ab = arow + kf * 64 + hi * 16;
        a.u[0] = *(const uint4*)(ab);
        a.u[1] = *(const uint4*)(ab + 32);
#pragma unroll
        for (int j = 0; j < 4; ++j) {
            const int   n  = n0 + j * 16 + lr;
            const char* bb = (const char*)wihh + (size_t)n * (EMBED * 2) + kf * 64 + hi * 16;
            FragU b;
            b.u[0] = *(const uint4*)(bb);
            b.u[1] = *(const uint4*)(bb + 32);
            acc[j] = wmma_f16(a.h, b.h, acc[j]);
        }
    }
#pragma unroll
    for (int j = 0; j < 4; ++j) {
        const int   n    = n0 + j * 16 + lr;
        const float bias = b_ih[n];
#pragma unroll
        for (int r = 0; r < 8; ++r) {           // C: VGPR r -> row r (lanes<16) / r+8 (lanes>=16)
            const int mm = m0 + r + hi * 8;
            xproj[(size_t)mm * HIDDEN + n] = acc[j][r] + bias;   // RT: stays L2-hot for the scan
        }
    }
}

// ---------------------------------------------------------------------------
// Kernel 2: sequential scan. ONE workgroup, 1024 threads = 32 waves.
// Wave w owns hidden cols [16w, 16w+16). Its 16x512 slice of W_hh^T lives in
// VGPRs as 16 pre-converted f16 B-fragments. h (padded to 16 rows, f16) is
// double-buffered in LDS; A-fragments are rebuilt each step with b128 DS loads.
// Also emits out_seq in packed-f16 GEMM-A layout for the logits kernel.
// ---------------------------------------------------------------------------
__global__ __launch_bounds__(1024, 1) void rnn_scan_kernel(
    const float* __restrict__ hidden0,  // [BATCH][HIDDEN]
    const float* __restrict__ W_hh,     // [HIDDEN][HIDDEN]
    const float* __restrict__ b_hh,     // [HIDDEN]
    const float* __restrict__ xproj,    // [M_TOTAL][HIDDEN]
    _Float16* __restrict__ outseq,      // [M_TOTAL][HIDDEN] f16
    float* __restrict__ hfinal)         // [BATCH][HIDDEN]
{
    __shared__ _Float16 hbuf[2][16 * HIDDEN];   // 2 x 16KB double buffer

    const int tid  = threadIdx.x;
    const int lane = tid & 31;
    const int wave = tid >> 5;                  // 0..31
    const int lr   = lane & 15;
    const int hi   = lane >> 4;
    const int n    = wave * 16 + lr;            // this lane's output column

    // init h buffer 0 (rows >= BATCH are zero padding; keeps A rows 8..15 zero)
    for (int i = tid; i < 16 * HIDDEN; i += 1024) {
        int   mrow = i / HIDDEN;
        int   col  = i % HIDDEN;
        float v    = (mrow < BATCH) ? hidden0[mrow * HIDDEN + col] : 0.0f;
        hbuf[0][i] = (_Float16)v;
    }

    // preload W_hh B-fragments: B(k,n) = W_hh[n][k]; lane column n, K swizzled:
    // lanes<16 -> K {0..7}u{16..23}, lanes>=16 -> K {8..15}u{24..31} (per 32-K fragment)
    v16h bfrag[16];
    {
        const float* wrow = W_hh + (size_t)n * HIDDEN;
#pragma unroll
        for (int kf = 0; kf < 16; ++kf) {
            const int kbase = kf * 32 + hi * 8;
            v16h f;
#pragma unroll
            for (int e = 0; e < 8; ++e) f[e] = (_Float16)wrow[kbase + e];
#pragma unroll
            for (int e = 0; e < 8; ++e) f[8 + e] = (_Float16)wrow[kbase + 16 + e];
            bfrag[kf] = f;
        }
    }
    const float bias = b_hh[n];
    __syncthreads();

    int cur = 0;
    for (int t = 0; t < SEQ; ++t) {
        // build A fragments from current h and multiply-accumulate over K=512
        v8f acc = {};
        const char* hb = (const char*)&hbuf[cur][0] + lr * (HIDDEN * 2) + hi * 16;
#pragma unroll
        for (int kf = 0; kf < 16; ++kf) {
            FragU a;
            a.u[0] = *(const uint4*)(hb + kf * 64);
            a.u[1] = *(const uint4*)(hb + kf * 64 + 32);
            acc = wmma_f16(a.h, bfrag[kf], acc);
        }

        const int  nxt = cur ^ 1;
        _Float16*  hw  = &hbuf[nxt][0];
#pragma unroll
        for (int r = 0; r < 8; ++r) {
            const int mrow = r + hi * 8;        // C row for this VGPR
            float v = 0.0f;
            if (mrow < BATCH) {
                v = tanhf(acc[r] + xproj[((size_t)mrow * SEQ + t) * HIDDEN + n] + bias);
                if (t + 1 < SEQ) {              // hide next step's x_proj latency
                    __builtin_prefetch(&xproj[((size_t)mrow * SEQ + (t + 1)) * HIDDEN + n], 0, 1);
                }
            }
            const _Float16 vh = (_Float16)v;
            hw[mrow * HIDDEN + n] = vh;         // padded rows stay 0
            if (mrow < BATCH) {
                outseq[((size_t)mrow * SEQ + t) * HIDDEN + n] = vh;
                if (t == SEQ - 1) hfinal[mrow * HIDDEN + n] = v;
            }
        }
        __syncthreads();
        cur = nxt;
    }
}

// ---------------------------------------------------------------------------
// Kernel 3: logits[m][v] = sum_k outseq[m][k] * W_out[v][k] + b_out[v]
// Block = 256 thr (8 waves); the shared 16x512 f16 A-tile is staged ONCE per
// block into LDS with async global->LDS b128 copies (ASYNCcnt), then each wave
// computes a 16x64 tile: A-fragments from LDS (ds_load_b128), B streamed from
// the L2-resident f16 W_out. Output stores are non-temporal (1.05 GB stream,
// never re-read -> keep W_out/out_seq hot in the 192 MB L2).
// ---------------------------------------------------------------------------
__global__ __launch_bounds__(256) void logits_kernel(
    const uint32_t* __restrict__ outseqh, // [M_TOTAL][HIDDEN/2] packed f16
    const uint32_t* __restrict__ wouth,   // [VOCAB][HIDDEN/2] packed f16
    const float* __restrict__ b_out,
    float* __restrict__ logits)           // [M_TOTAL][VOCAB] f32
{
    __shared__ __align__(16) _Float16 aTile[16 * HIDDEN];   // 16 KB

    const int tid  = threadIdx.x;
    const int lane = tid & 31;
    const int wave = tid >> 5;
    const int lr   = lane & 15;
    const int hi   = lane >> 4;
    const int m0   = blockIdx.y * 16;
    const int n64  = blockIdx.x * 8 + wave;

    // Stage A-tile (rows m0..m0+15 of out_seq, 16 KB) into LDS: 1024 b128
    // chunks, 4 per thread. All waves participate BEFORE any wave may exit.
    {
        const char* gsrc = (const char*)outseqh + (size_t)m0 * (HIDDEN * 2);
        char*       ldst = (char*)aTile;
#pragma unroll
        for (int j = 0; j < 4; ++j) {
            const int idx = tid + j * 256;      // chunk index; row-major contiguous
            copy16_g2lds(gsrc + idx * 16, ldst + idx * 16);
        }
        wait_async_copies();
        __syncthreads();
    }

    if (n64 < VOCAB / 64) {                     // wave-uniform: EXEC all-1s inside
        const int n0 = n64 * 64;
        const char* arow = (const char*)aTile + lr * (HIDDEN * 2);

        v8f acc[4] = {};
#pragma unroll
        for (int kf = 0; kf < HIDDEN / 32; ++kf) {
            FragU a;                            // A-fragment from LDS (ds_load_b128)
            const char* ab = arow + kf * 64 + hi * 16;
            a.u[0] = *(const uint4*)(ab);
            a.u[1] = *(const uint4*)(ab + 32);
#pragma unroll
            for (int j = 0; j < 4; ++j) {
                const int   v  = n0 + j * 16 + lr;
                const char* bb = (const char*)wouth + (size_t)v * (HIDDEN * 2) + kf * 64 + hi * 16;
                FragU b;
                b.u[0] = *(const uint4*)(bb);
                b.u[1] = *(const uint4*)(bb + 32);
                acc[j] = wmma_f16(a.h, b.h, acc[j]);
            }
        }
#pragma unroll
        for (int j = 0; j < 4; ++j) {
            const int   v    = n0 + j * 16 + lr;
            const float bias = b_out[v];
#pragma unroll
            for (int r = 0; r < 8; ++r) {
                const int mm = m0 + r + hi * 8;
                __builtin_nontemporal_store(acc[j][r] + bias,
                                            &logits[(size_t)mm * VOCAB + v]);
            }
        }
    }
}

// ---------------------------------------------------------------------------
// Workspace layout (bytes, all 256-aligned). Total ~75.2 MB.
// ---------------------------------------------------------------------------
#define OFF_EMBH   ((size_t)0)                              // 16000*128 u32 = 8,192,000
#define OFF_WIHH   ((size_t)8192000)                        //   512*128 u32 =   262,144
#define OFF_WOUTH  ((size_t)8454144)                        // 16000*256 u32 = 16,384,000
#define OFF_XPROJ  ((size_t)24838144)                       // 16384*512 f32 = 33,554,432
#define OFF_OUTSEQ ((size_t)58392576)                       // 16384*512 f16 = 16,777,216

extern "C" void kernel_launch(void* const* d_in, const int* in_sizes, int n_in,
                              void* d_out, int out_size, void* d_ws, size_t ws_size,
                              hipStream_t stream) {
    const int*   x     = (const int*)d_in[0];
    const float* hid0  = (const float*)d_in[1];
    const float* emb   = (const float*)d_in[2];
    const float* W_ih  = (const float*)d_in[3];
    const float* W_hh  = (const float*)d_in[4];
    const float* b_ih  = (const float*)d_in[5];
    const float* b_hh  = (const float*)d_in[6];
    const float* W_out = (const float*)d_in[7];
    const float* b_out = (const float*)d_in[8];

    char*     ws      = (char*)d_ws;
    uint32_t* embh    = (uint32_t*)(ws + OFF_EMBH);
    uint32_t* wihh    = (uint32_t*)(ws + OFF_WIHH);
    uint32_t* wouth   = (uint32_t*)(ws + OFF_WOUTH);
    float*    xproj   = (float*)(ws + OFF_XPROJ);
    _Float16* outseq  = (_Float16*)(ws + OFF_OUTSEQ);

    float* logits = (float*)d_out;                              // [16384][16000]
    float* hfinal = logits + (size_t)M_TOTAL * VOCAB;           // [8][512] tail

    // 0) f32 -> packed f16 weight conversions
    {
        int np;
        np = VOCAB * EMBED / 2;
        pack_f32_to_f16<<<(np + 255) / 256, 256, 0, stream>>>(emb, embh, np);
        np = HIDDEN * EMBED / 2;
        pack_f32_to_f16<<<(np + 255) / 256, 256, 0, stream>>>(W_ih, wihh, np);
        np = VOCAB * HIDDEN / 2;
        pack_f32_to_f16<<<(np + 255) / 256, 256, 0, stream>>>(W_out, wouth, np);
    }

    // 1) x_proj = embed(x) @ W_ih^T + b_ih   (1024 M-tiles x 8 waves)
    xproj_kernel<<<dim3(M_TOTAL / 16), 256, 0, stream>>>(x, embh, wihh, b_ih, xproj);

    // 2) sequential RNN scan: one workgroup, 32 waves, WMMA per step
    rnn_scan_kernel<<<dim3(1), 1024, 0, stream>>>(hid0, W_hh, b_hh, xproj, outseq, hfinal);

    // 3) logits = out_seq @ W_out^T + b_out  (dominant 268 GFLOP GEMM)
    logits_kernel<<<dim3(32, M_TOTAL / 16), 256, 0, stream>>>(
        (const uint32_t*)outseq, wouth, b_out, logits);
}
